// DilatedSelfAttention_30777735643242
// MI455X (gfx1250) — matrix operand order repaired and checked by reading, not compiled, not run on hardware
//
#include <hip/hip_runtime.h>
#include <stdint.h>

typedef __attribute__((ext_vector_type(16))) __bf16 bf16x16;
typedef __attribute__((ext_vector_type(8)))  float  f32x8;

#define B_     4
#define N_     8192
#define C_     128
#define S_     2048
#define KTOT_  14336
#define NBLK_  28
#define BLKPB_ 7
#define SCALE_ 0.08838834764831845f   // 1/sqrt(128)

union ABf { bf16x16 v; uint4 u4[2]; };
union Pk8 { __bf16 h[8]; uint4 u4; };

static __device__ __forceinline__ f32x8 zero8() {
  f32x8 z;
  #pragma unroll
  for (int i = 0; i < 8; ++i) z[i] = 0.0f;
  return z;
}

// Async global->LDS copy: 16 bytes per lane, tracked by ASYNCcnt.
static __device__ __forceinline__ void async_ld16(const void* g, void* l) {
  asm volatile("global_load_async_to_lds_b128 %0, %1, off"
               :: "v"((uint32_t)(uintptr_t)l), "v"((uint64_t)(uintptr_t)g)
               : "memory");
}

// ---------------------------------------------------------------------------
// Stage 1: gather + project Q/K/V.  Q,K row-major bf16; V stored transposed.
// grid = NBLK_*16 workgroups of 256 threads (8 waves); each does 128x128x128.
// ---------------------------------------------------------------------------
__global__ __launch_bounds__(256)
void proj_kernel(const float* __restrict__ x,
                 const float* __restrict__ Wq,
                 const float* __restrict__ Wk,
                 const float* __restrict__ Wv,
                 const int*   __restrict__ idx,
                 __bf16* __restrict__ Qb,
                 __bf16* __restrict__ Kb,
                 __bf16* __restrict__ Vt)
{
  constexpr int LDX = 136;                      // padded row stride (elems)
  __shared__ __align__(16) __bf16 Xs[128 * LDX];
  __shared__ __align__(16) __bf16 Ws[128 * LDX];

  const int bi   = blockIdx.x >> 4;             // sub-attention block 0..27
  const int tile = blockIdx.x & 15;             // 128-row tile within block
  const int b    = bi / BLKPB_;
  const int jb   = (bi % BLKPB_) * S_ + tile * 128;
  const int tid  = threadIdx.x;

  // gather 128 rows of x into LDS as bf16
  {
    const int row = tid >> 1;
    const int cb  = (tid & 1) * 64;
    const int g   = idx[jb + row];
    const float4* src = (const float4*)(x + ((size_t)b * N_ + g) * C_ + cb);
    __bf16* dst = &Xs[row * LDX + cb];
    #pragma unroll
    for (int i = 0; i < 16; ++i) {
      float4 f = src[i];
      dst[i*4+0] = (__bf16)f.x; dst[i*4+1] = (__bf16)f.y;
      dst[i*4+2] = (__bf16)f.z; dst[i*4+3] = (__bf16)f.w;
    }
  }

  const float* Wp[3] = { Wq, Wk, Wv };
  const int wave = tid >> 5, lane = tid & 31;
  const int hf = lane >> 4, l15 = lane & 15;

  for (int p = 0; p < 3; ++p) {
    __syncthreads();                            // Xs ready / prev pass done
    const float* W = Wp[p];
    #pragma unroll
    for (int i = 0; i < 64; ++i) {              // Ws[n][k] = W[k][n]
      int e = tid + i * 256;
      int k = e >> 7, n = e & 127;
      Ws[n * LDX + k] = (__bf16)W[e];
    }
    __syncthreads();

    // A fragments: wave's 16 rows, K=128 in 4 slices of 32
    ABf af[4];
    const int arow = wave * 16 + l15;
    #pragma unroll
    for (int ks = 0; ks < 4; ++ks) {
      af[ks].u4[0] = *(const uint4*)&Xs[arow * LDX + ks*32 + hf*8];
      af[ks].u4[1] = *(const uint4*)&Xs[arow * LDX + ks*32 + 16 + hf*8];
    }

    #pragma unroll
    for (int nt = 0; nt < 8; ++nt) {
      f32x8 acc = zero8();
      const int col = nt * 16 + l15;
      #pragma unroll
      for (int ks = 0; ks < 4; ++ks) {
        ABf bf;
        bf.u4[0] = *(const uint4*)&Ws[col * LDX + ks*32 + hf*8];
        bf.u4[1] = *(const uint4*)&Ws[col * LDX + ks*32 + 16 + hf*8];
        acc = __builtin_amdgcn_wmma_f32_16x16x32_bf16(false, af[ks].v, false, bf.v,
                                                      (short)0, acc, false, false);
      }
      if (p < 2) {                              // Q / K row-major
        __bf16* outp = (p == 0) ? Qb : Kb;
        const size_t rbase = (size_t)bi * S_ + tile * 128 + wave * 16 + 8 * hf;
        #pragma unroll
        for (int r = 0; r < 8; ++r)
          outp[(rbase + r) * C_ + col] = (__bf16)acc[r];
      } else {                                  // V transposed: Vt[bi][c][srow]
        Pk8 pk;
        #pragma unroll
        for (int r = 0; r < 8; ++r) pk.h[r] = (__bf16)acc[r];
        const size_t off = ((size_t)bi * C_ + col) * S_
                         + tile * 128 + wave * 16 + 8 * hf;
        *(uint4*)&Vt[off] = pk.u4;
      }
    }
  }
}

// ---------------------------------------------------------------------------
// Stage 2: causal flash attention per (block, 64-row query tile).
// 4 waves/WG; wave owns 16 query rows. K/V tiles (BN=32 keys) are staged
// cooperatively into double-buffered LDS via async global->LDS copies
// (ASYNCcnt), overlapping the copy of tile t+1 with WMMA work on tile t.
// Uniform tile count across waves; fully-masked tiles are inert in the
// online softmax. Emits un-normalized O and denom.
// ---------------------------------------------------------------------------
__global__ __launch_bounds__(128)
void attn_kernel(const __bf16* __restrict__ Qb,
                 const __bf16* __restrict__ Kb,
                 const __bf16* __restrict__ Vt,
                 float* __restrict__ Of,
                 float* __restrict__ Dn)
{
  constexpr int BM = 64, BN = 32, QT = S_ / BM;
  constexpr int LDK = 136;                      // padded K-tile row stride
  constexpr int LDV = 40;                       // padded V-tile row stride
  const int bi = blockIdx.x / QT;
  const int qt = blockIdx.x % QT;
  const int tid = threadIdx.x;
  const int wave = tid >> 5, lane = tid & 31;
  const int hf = lane >> 4, l15 = lane & 15;
  const int qrow0 = qt * BM + wave * 16;

  __shared__ __align__(16) __bf16 Ks[2][32 * LDK];   // keys x C (padded)
  __shared__ __align__(16) __bf16 Vs[2][128 * LDV];  // C x keys (padded)
  __shared__ __align__(16) __bf16 Plds[4][16][40];

  const __bf16* Qblk = Qb + (size_t)bi * S_ * C_;
  const __bf16* Kblk = Kb + (size_t)bi * S_ * C_;
  const __bf16* Vblk = Vt + (size_t)bi * C_ * S_;

  // stage tile t (keys [kv0, kv0+32)) into buffer buf: 8 async b128 / wave
  auto issue_tile = [&](int kv0, int buf) {
    #pragma unroll
    for (int i = 0; i < 4; ++i) {               // K tile: 32 rows x 256B
      const int chunk = i * 128 + tid;
      const int key = chunk >> 4, c16 = chunk & 15;
      async_ld16(Kblk + (size_t)(kv0 + key) * C_ + c16 * 8,
                 &Ks[buf][key * LDK + c16 * 8]);
    }
    #pragma unroll
    for (int i = 0; i < 4; ++i) {               // V^T tile: 128 rows x 64B
      const int chunk = i * 128 + tid;
      const int c = chunk >> 2, c4 = chunk & 3;
      async_ld16(Vblk + (size_t)c * S_ + kv0 + c4 * 8,
                 &Vs[buf][c * LDV + c4 * 8]);
    }
  };

  ABf qf[4];
  const int qrow = qrow0 + l15;
  #pragma unroll
  for (int ks = 0; ks < 4; ++ks) {
    qf[ks].u4[0] = *(const uint4*)&Qblk[qrow * C_ + ks*32 + hf*8];
    qf[ks].u4[1] = *(const uint4*)&Qblk[qrow * C_ + ks*32 + 16 + hf*8];
  }

  f32x8 o[8];
  #pragma unroll
  for (int c = 0; c < 8; ++c) o[c] = zero8();
  float mrow[8], lrow[8];
  #pragma unroll
  for (int r = 0; r < 8; ++r) { mrow[r] = -1e30f; lrow[r] = 0.0f; }

  const int ntiles = (qt * BM + BM) / BN;       // uniform across waves
  issue_tile(0, 0);

  for (int t = 0; t < ntiles; ++t) {
    const int kv0 = t * BN;
    const int buf = t & 1;
    if (t + 1 < ntiles) {
      issue_tile(kv0 + BN, buf ^ 1);
      asm volatile("s_wait_asynccnt 0x8" ::: "memory");  // tile t done
    } else {
      asm volatile("s_wait_asynccnt 0x0" ::: "memory");
    }
    __syncthreads();                            // all waves' copies visible

    f32x8 sacc0 = zero8(), sacc1 = zero8();
    #pragma unroll
    for (int ks = 0; ks < 4; ++ks) {
      ABf kf0, kf1;
      kf0.u4[0] = *(const uint4*)&Ks[buf][l15 * LDK + ks*32 + hf*8];
      kf0.u4[1] = *(const uint4*)&Ks[buf][l15 * LDK + ks*32 + 16 + hf*8];
      kf1.u4[0] = *(const uint4*)&Ks[buf][(16 + l15) * LDK + ks*32 + hf*8];
      kf1.u4[1] = *(const uint4*)&Ks[buf][(16 + l15) * LDK + ks*32 + 16 + hf*8];
      sacc0 = __builtin_amdgcn_wmma_f32_16x16x32_bf16(false, qf[ks].v, false, kf0.v,
                                                      (short)0, sacc0, false, false);
      sacc1 = __builtin_amdgcn_wmma_f32_16x16x32_bf16(false, qf[ks].v, false, kf1.v,
                                                      (short)0, sacc1, false, false);
    }

    // scale + causal mask; per-row (per-r) tile max over the 16-lane half
    float ev0[8], ev1[8], tm[8];
    #pragma unroll
    for (int r = 0; r < 8; ++r) {
      const int row = qrow0 + r + 8 * hf;
      const int c0 = kv0 + l15, c1 = kv0 + 16 + l15;
      float s0 = (c0 <= row) ? sacc0[r] * SCALE_ : -1e9f;
      float s1 = (c1 <= row) ? sacc1[r] * SCALE_ : -1e9f;
      ev0[r] = s0; ev1[r] = s1;
      tm[r] = fmaxf(s0, s1);
    }
    #pragma unroll
    for (int off = 1; off < 16; off <<= 1)
      #pragma unroll
      for (int r = 0; r < 8; ++r)
        tm[r] = fmaxf(tm[r], __shfl_xor(tm[r], off));

    float alpha[8], rs[8];
    #pragma unroll
    for (int r = 0; r < 8; ++r) {
      float mn = fmaxf(mrow[r], tm[r]);
      alpha[r] = __expf(mrow[r] - mn);
      mrow[r] = mn;
      float e0 = __expf(ev0[r] - mn);
      float e1 = __expf(ev1[r] - mn);
      ev0[r] = e0; ev1[r] = e1;
      rs[r] = e0 + e1;
    }
    #pragma unroll
    for (int off = 1; off < 16; off <<= 1)
      #pragma unroll
      for (int r = 0; r < 8; ++r)
        rs[r] += __shfl_xor(rs[r], off);
    #pragma unroll
    for (int r = 0; r < 8; ++r) lrow[r] = lrow[r] * alpha[r] + rs[r];
    #pragma unroll
    for (int c = 0; c < 8; ++c)
      #pragma unroll
      for (int r = 0; r < 8; ++r)
        o[c][r] *= alpha[r];

    // D-layout -> A-layout via per-wave LDS tile (same-wave LDS is in-order)
    #pragma unroll
    for (int r = 0; r < 8; ++r) {
      Plds[wave][r + 8*hf][l15]      = (__bf16)ev0[r];
      Plds[wave][r + 8*hf][16 + l15] = (__bf16)ev1[r];
    }
    asm volatile("s_wait_dscnt 0x0" ::: "memory");
    ABf pf;
    pf.u4[0] = *(const uint4*)&Plds[wave][l15][hf*8];
    pf.u4[1] = *(const uint4*)&Plds[wave][l15][16 + hf*8];

    #pragma unroll
    for (int c = 0; c < 8; ++c) {
      ABf vf;
      const int cc = c * 16 + l15;
      vf.u4[0] = *(const uint4*)&Vs[buf][cc * LDV + hf*8];
      vf.u4[1] = *(const uint4*)&Vs[buf][cc * LDV + 16 + hf*8];
      o[c] = __builtin_amdgcn_wmma_f32_16x16x32_bf16(false, pf.v, false, vf.v,
                                                     (short)0, o[c], false, false);
    }
    __syncthreads();   // all waves done reading buf before it is re-filled
  }

  const size_t row0 = (size_t)bi * S_ + qt * BM + wave * 16 + 8 * hf;
  #pragma unroll
  for (int c = 0; c < 8; ++c)
    #pragma unroll
    for (int r = 0; r < 8; ++r)
      Of[(row0 + r) * C_ + c * 16 + l15] = o[c][r];
  if (l15 == 0) {
    #pragma unroll
    for (int r = 0; r < 8; ++r) Dn[row0 + r] = lrow[r];
  }
}

// ---------------------------------------------------------------------------
// Stage 3: merge. Index structure is analytic: j0=n; j1 iff n odd; j2 iff n%4==1.
// out[b,n,:] = sum_j O_unnorm[b,j,:] / sum_j denom[b,j]   (pure gather)
// ---------------------------------------------------------------------------
__global__ __launch_bounds__(128)
void merge_kernel(const float* __restrict__ Of,
                  const float* __restrict__ Dn,
                  float* __restrict__ outp)
{
  const int b = blockIdx.x / N_;
  const int n = blockIdx.x % N_;
  const int c = threadIdx.x;
  const size_t base = (size_t)b * KTOT_;

  const int j0 = n;
  float d   = Dn[base + j0];
  float acc = Of[(base + j0) * C_ + c];
  if (n & 1) {
    const int j1 = 8192 + (n >> 12) * 2048 + (((n & 4095) - 1) >> 1);
    d   += Dn[base + j1];
    acc += Of[(base + j1) * C_ + c];
  }
  if ((n & 3) == 1) {
    const int j2 = 12288 + ((n - 1) >> 2);
    d   += Dn[base + j2];
    acc += Of[(base + j2) * C_ + c];
  }
  outp[((size_t)b * N_ + n) * C_ + c] = acc / d;
}

// ---------------------------------------------------------------------------
extern "C" void kernel_launch(void* const* d_in, const int* in_sizes, int n_in,
                              void* d_out, int out_size, void* d_ws, size_t ws_size,
                              hipStream_t stream) {
  const float* x  = (const float*)d_in[0];
  const float* Wq = (const float*)d_in[1];
  const float* Wk = (const float*)d_in[2];
  const float* Wv = (const float*)d_in[3];
  const int*   idx = (const int*)d_in[4];
  float* outp = (float*)d_out;

  const size_t R = (size_t)B_ * KTOT_;          // 57344 sparse rows
  __bf16* Qb = (__bf16*)d_ws;
  __bf16* Kb = Qb + R * C_;
  __bf16* Vt = Kb + R * C_;                     // NBLK_*C_*S_ == R*C_ elems
  float*  Of = (float*)(Vt + R * C_);
  float*  Dn = Of + R * C_;

  proj_kernel<<<NBLK_ * 16, 256, 0, stream>>>(x, Wq, Wk, Wv, idx, Qb, Kb, Vt);
  attn_kernel<<<NBLK_ * (S_ / 64), 128, 0, stream>>>(Qb, Kb, Vt, Of, Dn);
  merge_kernel<<<B_ * N_, 128, 0, stream>>>(Of, Dn, outp);
}
